// GraphColorizer_67130338837053
// MI455X (gfx1250) — compile-verified, matrix operand order checked by reading, not run on hardware
//
#include <hip/hip_runtime.h>
#include <hip/hip_bf16.h>

typedef __attribute__((ext_vector_type(16))) _Float16 v16h;
typedef __attribute__((ext_vector_type(8)))  float    v8f;

#define EMB 512
#define NC 20
#define HID 266
#define HIDP 288            // padded hidden, multiple of 32
#define ROWS_PER_BLOCK 128

#define KS1 16              // 512/32 k-steps, gemm1
#define OT1 32              // 512/16 col tiles, gemm1
#define KS2 16              // 512/32 k-steps, gemm2
#define OT2 18              // 288/16 col tiles, gemm2
#define KS3 9               // 288/32 k-steps, gemm3
#define OT3 2               // 32/16 col tiles, gemm3

// ws layout, in _Float16 elements
#define B1_ELEMS (KS1*OT1*32*16)           // 262144
#define B2_ELEMS (KS2*OT2*32*16)           // 147456
#define B3_ELEMS (KS3*OT3*32*16)           // 9216
#define B1_OFF 0
#define B2_OFF (B1_ELEMS)
#define B3_OFF (B1_ELEMS + B2_ELEMS)
#define PARTIAL_BYTE_OFF (1u<<20)          // block partial losses @ 1MB

#define WMMA_F16(A,B,C) \
  __builtin_amdgcn_wmma_f32_16x16x32_f16(false,(A),false,(B),(short)0,(C),false,false)

// CDNA5 16-bit A/B fragment position (ISA 7.12.2):
// lane = hi*16 + (row/col within tile); 16 halfs per lane indexed by j.
// kin -> (hi, j):  hi = bit3 of kin;  j = (kin&7) | ((kin>>4)<<3)
__device__ __forceinline__ void fragPos(int kin, int& hi, int& j) {
  hi = (kin >> 3) & 1;
  j  = (kin & 7) | ((kin >> 4) << 3);
}

// -------------------- weight repack: f32 -> f16 B-fragments --------------------
__global__ void prep_weights(const float* __restrict__ W1,
                             const float* __restrict__ W2,
                             const float* __restrict__ W3,
                             _Float16* __restrict__ wsH) {
  int idx = blockIdx.x * blockDim.x + threadIdx.x;
  const int total = B1_ELEMS + B2_ELEMS + B3_ELEMS;
  if (idx >= total) return;

  float val = 0.0f;
  int dst;
  if (idx < B1_ELEMS) {
    int t = idx;
    int j = t & 15, lane = (t >> 4) & 31, rem = t >> 9;
    int ot = rem % OT1, ks = rem / OT1;
    int hi = lane >> 4, n = lane & 15;
    int kin = (j & 7) + (hi << 3) + ((j >> 3) << 4);
    int k = ks * 32 + kin, o = ot * 16 + n;            // o<512, k<512
    val = W1[o * EMB + k];
    dst = B1_OFF + t;
  } else if (idx < B1_ELEMS + B2_ELEMS) {
    int t = idx - B1_ELEMS;
    int j = t & 15, lane = (t >> 4) & 31, rem = t >> 9;
    int ot = rem % OT2, ks = rem / OT2;
    int hi = lane >> 4, n = lane & 15;
    int kin = (j & 7) + (hi << 3) + ((j >> 3) << 4);
    int k = ks * 32 + kin, o = ot * 16 + n;            // o<288, k<512
    val = (o < HID) ? W2[o * EMB + k] : 0.0f;
    dst = B2_OFF + t;
  } else {
    int t = idx - (B1_ELEMS + B2_ELEMS);
    int j = t & 15, lane = (t >> 4) & 31, rem = t >> 9;
    int ot = rem % OT3, ks = rem / OT3;
    int hi = lane >> 4, n = lane & 15;
    int kin = (j & 7) + (hi << 3) + ((j >> 3) << 4);
    int k = ks * 32 + kin, o = ot * 16 + n;            // k<288, o<32
    val = (o < NC && k < HID) ? W3[o * HID + k] : 0.0f;
    dst = B3_OFF + t;
  }
  wsH[dst] = (_Float16)val;
}

// -------------------- gemm2 chunk helper --------------------
template <int WDT>
__device__ __forceinline__ void gemm2_chunk(const _Float16* ldsB, _Float16* ldsA,
                                            const _Float16* __restrict__ B2,
                                            const float* __restrict__ b2,
                                            int w, int lane, int hi, int n, int og) {
  v8f acc[WDT];
#pragma unroll
  for (int i = 0; i < WDT; ++i) acc[i] = (v8f){};
#pragma unroll
  for (int ks = 0; ks < KS2; ++ks) {
    v16h a = *(const v16h*)(ldsB + ((w * 16 + ks) * 32 + lane) * 16);
#pragma unroll
    for (int i = 0; i < WDT; ++i) {
      v16h bb = *(const v16h*)(B2 + ((ks * OT2 + og + i) * 32 + lane) * 16);
      acc[i] = WMMA_F16(a, bb, acc[i]);
    }
  }
#pragma unroll
  for (int i = 0; i < WDT; ++i) {
    int col = (og + i) * 16 + n;
    float bias = (col < HID) ? b2[col] : 0.0f;
#pragma unroll
    for (int r = 0; r < 8; ++r) {
      int m = r + hi * 8;
      float val = acc[i][r] + bias;
      val = (val >= 0.0f) ? val : 0.01f * val;          // LeakyReLU(0.01)
      int kin = col & 31, ks3 = col >> 5;
      int fh, fj; fragPos(kin, fh, fj);
      ldsA[((w * 16 + ks3) * 32 + (fh * 16 + m)) * 16 + fj] = (_Float16)val;
    }
  }
}

// -------------------- fused 3-layer MLP + softmax + sample + loss --------------------
__global__ void __launch_bounds__(256, 1)
fused_mlp(const float* __restrict__ E, const float* __restrict__ b1,
          const float* __restrict__ b2, const float* __restrict__ b3,
          const _Float16* __restrict__ wsH,
          const int* __restrict__ vorder, const int* __restrict__ optc,
          float* __restrict__ out, float* __restrict__ partial, int Nrows) {
  extern __shared__ char smem[];
  _Float16* ldsA = (_Float16*)smem;                    // 131072 B : E-frags, later H2-frags
  _Float16* ldsB = (_Float16*)(smem + 131072);         // 131072 B : H1-frags
  float*    lBuf = (float*)(smem + 262144);            // 16384 B  : logits [128][32]
  float*    sArr = (float*)(smem + 278528);            // 1024 B   : loss reduce

  const int tid  = threadIdx.x;
  const int lane = tid & 31;
  const int w    = tid >> 5;       // wave id 0..7, owns rows [16w,16w+16)
  const int hi   = lane >> 4;
  const int n    = lane & 15;
  const int blockRow = blockIdx.x * ROWS_PER_BLOCK;

  // ---- stage E tile (128x512 f32 -> f16 A-fragment layout in LDS) ----
  for (int it = tid; it < ROWS_PER_BLOCK * EMB / 4; it += 256) {
    int row = it >> 7;                 // / (512/4)
    int kq  = (it & 127) << 2;
    int gr = blockRow + row; if (gr >= Nrows) gr = Nrows - 1;
    const float4 v = *(const float4*)(E + (size_t)gr * EMB + kq);
    int ks = kq >> 5, kin = kq & 31;
    int fh, fj; fragPos(kin, fh, fj);  // kq%4==0 -> fj%4==0, 4 halfs contiguous
    int lane_ = fh * 16 + (row & 15);
    int off = (((row >> 4) * 16 + ks) * 32 + lane_) * 16 + fj;
    union { uint2 u; _Float16 h[4]; } pk;
    pk.h[0] = (_Float16)v.x; pk.h[1] = (_Float16)v.y;
    pk.h[2] = (_Float16)v.z; pk.h[3] = (_Float16)v.w;
    *(uint2*)(ldsA + off) = pk.u;
  }
  __syncthreads();

  const _Float16* B1 = wsH + B1_OFF;
  const _Float16* B2 = wsH + B2_OFF;
  const _Float16* B3 = wsH + B3_OFF;

  // ---- GEMM1: H1[128,512] = leaky(E @ W1^T + b1), result to ldsB as A-frags ----
#pragma unroll 1
  for (int ct = 0; ct < 8; ++ct) {                     // 64-col tiles
    v8f acc[4];
#pragma unroll
    for (int i = 0; i < 4; ++i) acc[i] = (v8f){};
#pragma unroll
    for (int ks = 0; ks < KS1; ++ks) {
      v16h a = *(const v16h*)(ldsA + ((w * 16 + ks) * 32 + lane) * 16);
#pragma unroll
      for (int i = 0; i < 4; ++i) {
        v16h bb = *(const v16h*)(B1 + ((ks * OT1 + ct * 4 + i) * 32 + lane) * 16);
        acc[i] = WMMA_F16(a, bb, acc[i]);
      }
    }
#pragma unroll
    for (int i = 0; i < 4; ++i) {
      int col = ct * 64 + i * 16 + n;
      float bias = b1[col];
#pragma unroll
      for (int r = 0; r < 8; ++r) {
        int m = r + hi * 8;                            // D layout: vgpr r, hi half
        float val = acc[i][r] + bias;
        val = (val >= 0.0f) ? val : 0.01f * val;
        int kin = col & 31, ks2 = col >> 5;
        int fh, fj; fragPos(kin, fh, fj);
        ldsB[((w * 16 + ks2) * 32 + (fh * 16 + m)) * 16 + fj] = (_Float16)val;
      }
    }
  }

  // ---- GEMM2: H2[128,288] = leaky(H1 @ W2p^T + b2p), result to ldsA (reuse) ----
#pragma unroll 1
  for (int og = 0; og < 16; og += 4)
    gemm2_chunk<4>(ldsB, ldsA, B2, b2, w, lane, hi, n, og);
  gemm2_chunk<2>(ldsB, ldsA, B2, b2, w, lane, hi, n, 16);

  // ---- GEMM3: logits[128,32] = H2 @ W3p^T + b3p ----
#pragma unroll 1
  for (int ot = 0; ot < OT3; ++ot) {
    v8f acc = (v8f){};
#pragma unroll
    for (int ks = 0; ks < KS3; ++ks) {
      v16h a  = *(const v16h*)(ldsA + ((w * 16 + ks) * 32 + lane) * 16);
      v16h bb = *(const v16h*)(B3 + ((ks * OT3 + ot) * 32 + lane) * 16);
      acc = WMMA_F16(a, bb, acc);
    }
    int col = ot * 16 + n;
    float bias = (col < NC) ? b3[col] : 0.0f;
#pragma unroll
    for (int r = 0; r < 8; ++r) {
      int m = r + hi * 8;
      lBuf[(w * 16 + m) * 32 + col] = acc[r] + bias;
    }
  }
  __syncthreads();

  // ---- softmax + categorical (Gumbel-max) + per-vertex loss ----
  float myloss = 0.0f;
  if (tid < ROWS_PER_BLOCK) {
    int row = blockRow + tid;
    if (row < Nrows) {
      float lg[NC];
      float mx = -1e30f;
#pragma unroll
      for (int c = 0; c < NC; ++c) { lg[c] = lBuf[tid * 32 + c]; mx = fmaxf(mx, lg[c]); }
      float s = 0.0f;
#pragma unroll
      for (int c = 0; c < NC; ++c) s += __expf(lg[c] - mx);

      int best = 0; float bv = -1e30f;
#pragma unroll
      for (int c = 0; c < NC; ++c) {
        unsigned x = (unsigned)(row * NC + c);        // PCG-style hash -> U(0,1)
        x = x * 747796405u + 2891336453u;
        unsigned wd = ((x >> ((x >> 28) + 4u)) ^ x) * 277803737u;
        wd = (wd >> 22) ^ wd;
        float u = ((wd & 0xFFFFFFu) + 0.5f) * (1.0f / 16777216.0f);
        float g = -__logf(-__logf(u));
        float vc = lg[c] + g;
        if (vc > bv) { bv = vc; best = c; }
      }
      int v0 = vorder[0];
      out[row] = (row == v0) ? 0.0f : (float)best;

      int t = optc[row];
      float pt = __expf(lg[t] - mx) / s;
      float l = 1.0f - __logf(pt + 1e-10f);
      if (row != v0) myloss = l;                       // sum over vertex_order[1:]
    }
  }
  // deterministic block reduction of loss
  sArr[tid] = myloss;
  __syncthreads();
  for (int st = 128; st > 0; st >>= 1) {
    if (tid < st) sArr[tid] += sArr[tid + st];
    __syncthreads();
  }
  if (tid == 0) partial[blockIdx.x] = sArr[0];
}

// -------------------- deterministic final loss reduction --------------------
__global__ void reduce_loss(const float* __restrict__ partial, int nb,
                            float* __restrict__ out, int Nrows) {
  __shared__ float sh[256];
  int tid = threadIdx.x;
  float s = 0.0f;
  for (int i = tid; i < nb; i += 256) s += partial[i];
  sh[tid] = s;
  __syncthreads();
  for (int k = 128; k > 0; k >>= 1) {
    if (tid < k) sh[tid] += sh[tid + k];
    __syncthreads();
  }
  if (tid == 0) out[Nrows] = sh[0];
}

extern "C" void kernel_launch(void* const* d_in, const int* in_sizes, int n_in,
                              void* d_out, int out_size, void* d_ws, size_t ws_size,
                              hipStream_t stream) {
  const float* E      = (const float*)d_in[0];
  const float* W1     = (const float*)d_in[1];
  const float* b1     = (const float*)d_in[2];
  const float* W2     = (const float*)d_in[3];
  const float* b2     = (const float*)d_in[4];
  const float* W3     = (const float*)d_in[5];
  const float* b3     = (const float*)d_in[6];
  const int*   vorder = (const int*)d_in[7];
  const int*   optc   = (const int*)d_in[8];
  float* out = (float*)d_out;

  int Nrows = in_sizes[0] / EMB;
  _Float16* wsH  = (_Float16*)d_ws;
  float* partial = (float*)((char*)d_ws + PARTIAL_BYTE_OFF);

  const int totalPrep = B1_ELEMS + B2_ELEMS + B3_ELEMS;
  prep_weights<<<(totalPrep + 255) / 256, 256, 0, stream>>>(W1, W2, W3, wsH);

  int nb = (Nrows + ROWS_PER_BLOCK - 1) / ROWS_PER_BLOCK;
  size_t smemBytes = 131072 + 131072 + 16384 + 1024;   // 279552 B < 320KB/WGP
  fused_mlp<<<nb, 256, smemBytes, stream>>>(E, b1, b2, b3, wsH, vorder, optc,
                                            out, partial, Nrows);
  reduce_loss<<<1, 256, 0, stream>>>(partial, nb, out, Nrows);
}
